// GCN_63359357550603
// MI455X (gfx1250) — matrix operand order repaired and checked by reading, not compiled, not run on hardware
//
#include <hip/hip_runtime.h>
#include <hip/hip_bf16.h>

typedef __attribute__((ext_vector_type(16))) _Float16 v16h;
typedef __attribute__((ext_vector_type(8)))  float    v8f;

#define KDIM 128   // inner dimension is 128 for all three layers

// ---------------------------------------------------------------------------
// Degree / normalization kernels
// ---------------------------------------------------------------------------
__global__ void gcn_zero_f32(float* __restrict__ p, int n) {
    int i = blockIdx.x * blockDim.x + threadIdx.x;
    if (i < n) p[i] = 0.0f;
}

__global__ void gcn_deg_count(const int* __restrict__ dst, float* __restrict__ deg,
                              int n_edges) {
    int e = blockIdx.x * blockDim.x + threadIdx.x;
    if (e < n_edges) atomicAdd(&deg[dst[e]], 1.0f);
}

__global__ void gcn_make_dinv(float* __restrict__ deg, int n) {
    int i = blockIdx.x * blockDim.x + threadIdx.x;
    if (i < n) {
        float d = deg[i] + 1.0f;   // + self-loop
        deg[i] = rsqrtf(d);        // d >= 1 always
    }
}

// ---------------------------------------------------------------------------
// WMMA GEMM (K = 128 hardcoded, fully unrolled -> 4 v_wmma) with fused
// self-loop init:
//   XW[n_rows x n_cols]      = (relu?)H[n_rows x 128] * W[128 x n_cols]
//   outInit[n_rows x n_cols] = bias + dinv^2 * XW        (self-loop term)
// block: 256 threads (8 waves), tile 128 rows x 16 cols; each wave -> 16x16.
// ---------------------------------------------------------------------------
__global__ __launch_bounds__(256)
void gcn_gemm_wmma(const float* __restrict__ H, const float* __restrict__ W,
                   float* __restrict__ XW,
                   const float* __restrict__ dinv, const float* __restrict__ bias,
                   float* __restrict__ outInit,
                   int n_rows, int n_cols, int apply_relu) {
    __shared__ float Hs[128][36];  // 128 rows x 32-wide K chunk, stride 144B (16B aligned)
    __shared__ float Ws[32][17];   // 32 K x 16 cols (+1 pad)

    const int row0 = blockIdx.x * 128;
    const int col0 = blockIdx.y * 16;
    const int tid  = threadIdx.x;
    const int wave = tid >> 5;
    const int lane = tid & 31;
    const int m    = lane & 15;
    const int g    = lane >> 4;
    const bool fullBlock = (row0 + 128) <= n_rows;

    v8f acc = {};

#pragma unroll
    for (int kc = 0; kc < KDIM; kc += 32) {
        // ---- Stage H chunk as float4 (global_load_b128, coalesced rows) ----
        if (fullBlock) {
#pragma unroll
            for (int i = tid; i < 128 * 8; i += 256) {
                int r  = i >> 3;
                int c4 = (i & 7) << 2;
                float4 v = *(const float4*)(H + (size_t)(row0 + r) * KDIM + kc + c4);
                if (apply_relu) {
                    v.x = fmaxf(v.x, 0.0f); v.y = fmaxf(v.y, 0.0f);
                    v.z = fmaxf(v.z, 0.0f); v.w = fmaxf(v.w, 0.0f);
                }
                *(float4*)(&Hs[r][c4]) = v;
            }
        } else {
#pragma unroll
            for (int i = tid; i < 128 * 8; i += 256) {
                int r  = i >> 3;
                int c4 = (i & 7) << 2;
                float4 v = make_float4(0.0f, 0.0f, 0.0f, 0.0f);
                if (row0 + r < n_rows)
                    v = *(const float4*)(H + (size_t)(row0 + r) * KDIM + kc + c4);
                if (apply_relu) {
                    v.x = fmaxf(v.x, 0.0f); v.y = fmaxf(v.y, 0.0f);
                    v.z = fmaxf(v.z, 0.0f); v.w = fmaxf(v.w, 0.0f);
                }
                *(float4*)(&Hs[r][c4]) = v;
            }
        }
        // ---- Stage W chunk (small: 512 floats) ----
#pragma unroll
        for (int i = tid; i < 32 * 16; i += 256) {
            int r = i >> 4, c = i & 15;
            Ws[r][c] = W[(size_t)(kc + r) * n_cols + (col0 + c)];
        }
        __syncthreads();

        // Pack A (16x32 f16) per ISA 7.12.2: lanes 0-15 K=0..7(+16), 16-31 +8
        v16h a, b;
        const int wr = wave * 16 + m;
#pragma unroll
        for (int e = 0; e < 16; ++e) {
            int v = e >> 1, s = e & 1;
            int ka = ((v & 3) << 1) + s + (g << 3) + ((v >> 2) << 4);
            a[e] = (_Float16)Hs[wr][ka];
        }
        // Pack B (32x16 f16): lane holds column lane&15; same K order as A: K=0..7(+16), 16-31 +8
        const int nn = lane & 15;
        const int kb = g << 3;
#pragma unroll
        for (int e = 0; e < 16; ++e) {
            b[e] = (_Float16)Ws[kb + (e & 7) + ((e >> 3) << 4)][nn];
        }

        acc = __builtin_amdgcn_wmma_f32_16x16x32_f16(
            /*neg_a=*/false, a, /*neg_b=*/false, b,
            /*c_mod=*/(short)0, acc, /*reuse_a=*/false, /*reuse_b=*/false);
        __syncthreads();
    }

    // ---- Epilogue: D layout VGPR r -> M = r + g*8, N = lane&15.
    // Write XW and fused self-loop init  out = bias + dinv^2 * xw.
    const int nn = lane & 15;
    const float bcol = bias[col0 + nn];
    if (fullBlock) {
#pragma unroll
        for (int r = 0; r < 8; ++r) {
            int grow = row0 + wave * 16 + r + (g << 3);
            size_t idx = (size_t)grow * n_cols + (col0 + nn);
            float v  = acc[r];
            float di = dinv[grow];
            XW[idx]      = v;
            outInit[idx] = bcol + di * di * v;
        }
    } else {
#pragma unroll
        for (int r = 0; r < 8; ++r) {
            int grow = row0 + wave * 16 + r + (g << 3);
            if (grow < n_rows) {
                size_t idx = (size_t)grow * n_cols + (col0 + nn);
                float v  = acc[r];
                float di = dinv[grow];
                XW[idx]      = v;
                outInit[idx] = bcol + di * di * v;
            }
        }
    }
}

// ---------------------------------------------------------------------------
// Edge aggregation, float4-vectorized:
//   out[dst[e]] += dinv[src]*dinv[dst] * xw[src[e]]
// One thread per (edge, 4-channel group); C=128 -> one wave per edge,
// global_load_b128 row gather, coalesced scalar f32 atomics.
// cshift4 = log2(C/4)
// ---------------------------------------------------------------------------
__global__ void gcn_agg_edges(const float4* __restrict__ xw,
                              const float* __restrict__ dinv,
                              const int* __restrict__ src,
                              const int* __restrict__ dst,
                              float* __restrict__ out,
                              int n_edges, int cshift4) {
    unsigned long long t = (unsigned long long)blockIdx.x * blockDim.x + threadIdx.x;
    unsigned long long total = (unsigned long long)n_edges << cshift4;
    if (t >= total) return;
    int e  = (int)(t >> cshift4);
    int cg = (int)(t & ((1u << cshift4) - 1u));
    int s = src[e];
    int d = dst[e];
    float w = dinv[s] * dinv[d];
    float4 v = xw[((size_t)s << cshift4) + cg];
    float* o = out + ((((size_t)d << cshift4) + cg) << 2);
    atomicAdd(o + 0, w * v.x);
    atomicAdd(o + 1, w * v.y);
    atomicAdd(o + 2, w * v.z);
    atomicAdd(o + 3, w * v.w);
}

// ---------------------------------------------------------------------------
// log_softmax over 64 channels, one wave32 per node (2 channels per lane)
// ---------------------------------------------------------------------------
__global__ void gcn_log_softmax64(float* __restrict__ out, int n_nodes) {
    int node = blockIdx.x * (blockDim.x >> 5) + (threadIdx.x >> 5);
    int lane = threadIdx.x & 31;
    if (node >= n_nodes) return;
    float* row = out + (size_t)node * 64;
    float v0 = row[lane];
    float v1 = row[lane + 32];
    float mx = fmaxf(v0, v1);
#pragma unroll
    for (int o = 16; o > 0; o >>= 1) mx = fmaxf(mx, __shfl_xor(mx, o, 32));
    float s = __expf(v0 - mx) + __expf(v1 - mx);
#pragma unroll
    for (int o = 16; o > 0; o >>= 1) s += __shfl_xor(s, o, 32);
    float lse = mx + logf(s);
    row[lane]      = v0 - lse;
    row[lane + 32] = v1 - lse;
}

// ---------------------------------------------------------------------------
// Host side
// ---------------------------------------------------------------------------
extern "C" void kernel_launch(void* const* d_in, const int* in_sizes, int n_in,
                              void* d_out, int out_size, void* d_ws, size_t ws_size,
                              hipStream_t stream) {
    const int IN_CH = 128, HID_CH = 128, OUT_CH = 64;

    const float* x   = (const float*)d_in[0];
    const int* eidx  = (const int*)d_in[1];
    const float* W1  = (const float*)d_in[2];
    const float* b1  = (const float*)d_in[3];
    const float* W2  = (const float*)d_in[4];
    const float* b2  = (const float*)d_in[5];
    const float* W3  = (const float*)d_in[6];
    const float* b3  = (const float*)d_in[7];

    const int N = in_sizes[0] / IN_CH;       // 100000
    const int E = in_sizes[1] / 2;           // 1600000
    const int* src = eidx;
    const int* dst = eidx + E;

    // Workspace layout: dinv | bufA (N x 128) | bufB (N x 128)
    char* ws = (char*)d_ws;
    float* dinv = (float*)ws;
    size_t off  = (((size_t)N * sizeof(float)) + 255) & ~(size_t)255;
    float* bufA = (float*)(ws + off);
    float* bufB = bufA + (size_t)N * HID_CH;
    float* outp = (float*)d_out;             // N x 64

    const int T = 256;
    dim3 blk(T);

    // 1) degrees -> dinv
    gcn_zero_f32<<<dim3((N + T - 1) / T), blk, 0, stream>>>(dinv, N);
    gcn_deg_count<<<dim3((E + T - 1) / T), blk, 0, stream>>>(dst, dinv, E);
    gcn_make_dinv<<<dim3((N + T - 1) / T), blk, 0, stream>>>(dinv, N);

    const int rowBlocks = (N + 127) / 128;
    unsigned long long tE32 = (unsigned long long)E * 32ull;  // C=128: E * C/4
    unsigned long long tE16 = (unsigned long long)E * 16ull;  // C=64

    // ---- Layer 1: bufA = x@W1 ; bufB = b1 + dinv^2*bufA ; edges -> bufB
    gcn_gemm_wmma<<<dim3(rowBlocks, HID_CH / 16), blk, 0, stream>>>(
        x, W1, bufA, dinv, b1, bufB, N, HID_CH, /*relu=*/0);
    gcn_agg_edges<<<dim3((unsigned)((tE32 + T - 1) / T)), blk, 0, stream>>>(
        (const float4*)bufA, dinv, src, dst, bufB, E, 5);

    // ---- Layer 2: bufA = relu(bufB)@W2 ; bufB = b2 + dinv^2*bufA ; edges
    gcn_gemm_wmma<<<dim3(rowBlocks, HID_CH / 16), blk, 0, stream>>>(
        bufB, W2, bufA, dinv, b2, bufB, N, HID_CH, /*relu=*/1);
    gcn_agg_edges<<<dim3((unsigned)((tE32 + T - 1) / T)), blk, 0, stream>>>(
        (const float4*)bufA, dinv, src, dst, bufB, E, 5);

    // ---- Layer 3: bufA = relu(bufB)@W3 ; d_out = b3 + dinv^2*bufA ; edges
    gcn_gemm_wmma<<<dim3(rowBlocks, OUT_CH / 16), blk, 0, stream>>>(
        bufB, W3, bufA, dinv, b3, outp, N, OUT_CH, /*relu=*/1);
    gcn_agg_edges<<<dim3((unsigned)((tE16 + T - 1) / T)), blk, 0, stream>>>(
        (const float4*)bufA, dinv, src, dst, outp, E, 4);

    // ---- log_softmax over 64 channels, wave per node
    int wavesPerBlk = T / 32;
    gcn_log_softmax64<<<dim3((N + wavesPerBlk - 1) / wavesPerBlk), blk, 0, stream>>>(
        outp, N);
}
